// Attention_17849884082851
// MI455X (gfx1250) — matrix operand order repaired and checked by reading, not compiled, not run on hardware
//
#include <hip/hip_runtime.h>
#include <math.h>

typedef __attribute__((ext_vector_type(16))) _Float16 v16h;
typedef __attribute__((ext_vector_type(8)))  float    v8f;

#define BB 4
#define TT 2048
#define EE 64
#define HH 8
#define QQL 6
#define HE 512          // HH*EE
#define SROW 2080       // padded LDS row stride (floats) to rotate banks

// ---------------- QK causal conv projection -> f16 q (pre-scaled 1/sqrt(E)) and k ----
__global__ void qk_conv_kernel(const float* __restrict__ x, const float* __restrict__ Wqk,
                               const float* __restrict__ bqk,
                               _Float16* __restrict__ qh, _Float16* __restrict__ kh) {
    int idx = blockIdx.x * blockDim.x + threadIdx.x;      // over B*T*2HE
    if (idx >= BB * TT * 2 * HE) return;
    int o = idx % (2 * HE);
    int t = (idx / (2 * HE)) % TT;
    int b = idx / (2 * HE * TT);
    const float* w = Wqk + (size_t)o * (EE * QQL);
    float acc = bqk[o];
    int j0 = (t >= QQL - 1) ? 0 : (QQL - 1 - t);
    for (int j = j0; j < QQL; ++j) {
        const float* xr = x + ((size_t)b * TT + (t - (QQL - 1) + j)) * EE;
        #pragma unroll
        for (int i = 0; i < EE; ++i) acc = fmaf(xr[i], w[i * QQL + j], acc);
    }
    if (o < HE) {
        int hh = o >> 6, e = o & 63;
        qh[(((size_t)b * HH + hh) * TT + t) * EE + e] = (_Float16)(acc * 0.125f);
    } else {
        int oo = o - HE; int hh = oo >> 6, e = oo & 63;
        kh[(((size_t)b * HH + hh) * TT + t) * EE + e] = (_Float16)acc;
    }
}

// ------------- value projection -> f16 V stored TRANSPOSED [B,H,E,T] ----------------
// Transposed layout makes the P@V WMMA B-operand (lane = column e, 16 consecutive
// K tokens) a contiguous 32-byte read -> global_load_b128 instead of 16 strided d16.
__global__ void value_kernel(const float* __restrict__ x, const float* __restrict__ Wv,
                             const float* __restrict__ bv, _Float16* __restrict__ vt) {
    int idx = blockIdx.x * blockDim.x + threadIdx.x;      // over B*T*HE
    if (idx >= BB * TT * HE) return;
    int o = idx % HE;
    size_t bt = idx / HE;
    int t = (int)(bt % TT); int b = (int)(bt / TT);
    const float* xr = x + bt * EE;
    float acc = bv[o];
    #pragma unroll 8
    for (int i = 0; i < EE; ++i) acc = fmaf(xr[i], Wv[(size_t)i * HE + o], acc);
    int hh = o >> 6, e = o & 63;
    vt[(((size_t)b * HH + hh) * EE + e) * TT + t] = (_Float16)acc;
}

// ---------------- attention: scores (WMMA) + softmax + P@V (WMMA) -------------------
__global__ __launch_bounds__(256) void attn_kernel(
        const _Float16* __restrict__ qh, const _Float16* __restrict__ kh,
        const _Float16* __restrict__ vt, const unsigned char* __restrict__ mask,
        float* __restrict__ attn_out,   // [B,H,T,T]
        float* __restrict__ ctx) {      // [B,T,HE]
    extern __shared__ float smem[];     // 16 x SROW f32 score stripe (~130 KB, CDNA5 WGP LDS)
    int blk = blockIdx.x;
    int qt = blk % (TT / 16);
    int hh = (blk / (TT / 16)) % HH;
    int b  = blk / ((TT / 16) * HH);
    int q0 = qt * 16;
    int tid  = threadIdx.x;
    int wave = tid >> 5;
    int lane = tid & 31;
    int m  = lane & 15;
    int hi = lane >> 4;                 // half-wave select per WMMA lane layout

    const _Float16* qbase = qh + (((size_t)b * HH + hh) * TT) * EE;
    const _Float16* kbase = kh + (((size_t)b * HH + hh) * TT) * EE;
    const _Float16* vbase = vt + (((size_t)b * HH + hh) * EE) * TT;   // [E, T]

    // A operand: Q tile rows q0..q0+15, K=0..63 as two k=32 chunks.
    // k order chosen here (contiguous): lanes 0-15 hold K kb..kb+15 with kb=0; lanes 16-31 kb=16 (matches B).
    v16h a0, a1;
    {
        const _Float16* qr = qbase + (size_t)(q0 + m) * EE;
        int kb = hi * 16;
        #pragma unroll
        for (int i = 0; i < 8; ++i) {
            a0[i]     = qr[kb + i];
            a0[8 + i] = qr[kb + 8 + i];
            a1[i]     = qr[32 + kb + i];
            a1[8 + i] = qr[32 + kb + 8 + i];
        }
    }

    // ---- S = Q K^T (scale folded into qh), mask, stash f32 scores in LDS ----
    for (int ct = wave; ct < TT / 16; ct += 8) {
        int col = ct * 16 + m;
        const _Float16* kr = kbase + (size_t)col * EE;
        v16h b0, b1;
        int kb = hi * 16;               // B k order chosen here, same as A: lanes 0-15 K=0-15, lanes 16-31 K=16-31
        #pragma unroll
        for (int i = 0; i < 16; ++i) {
            b0[i] = kr[kb + i];
            b1[i] = kr[32 + kb + i];
        }
        v8f acc = {};
        acc = __builtin_amdgcn_wmma_f32_16x16x32_f16(false, a0, false, b0, (short)0, acc, false, false);
        acc = __builtin_amdgcn_wmma_f32_16x16x32_f16(false, a1, false, b1, (short)0, acc, false, false);
        #pragma unroll
        for (int r = 0; r < 8; ++r) {   // D layout: VGPR r -> row r (lanes 0-15) / r+8 (lanes 16-31)
            int row = r + hi * 8;
            unsigned char mk = mask[(size_t)(q0 + row) * TT + col];
            smem[row * SROW + col] = mk ? acc[r] : -1e9f;
        }
    }
    __syncthreads();

    // ---- row softmax: wave w owns rows 2w, 2w+1; wave-local shuffle reductions ----
    for (int rr = 0; rr < 2; ++rr) {
        int row = wave * 2 + rr;
        float* srow = smem + row * SROW;
        float mx = -1e30f;
        for (int c = lane; c < TT; c += 32) mx = fmaxf(mx, srow[c]);
        #pragma unroll
        for (int s = 16; s > 0; s >>= 1) mx = fmaxf(mx, __shfl_xor(mx, s, 32));
        float sum = 0.f;
        for (int c = lane; c < TT; c += 32) {
            float e = __expf(srow[c] - mx);
            srow[c] = e;
            sum += e;
        }
        #pragma unroll
        for (int s = 16; s > 0; s >>= 1) sum += __shfl_xor(sum, s, 32);
        float inv = 1.0f / sum;
        for (int c = lane; c < TT; c += 32) srow[c] *= inv;
    }
    __syncthreads();

    // ---- stream normalized weights to global (dominant 537MB output) ----
    {
        float* outb = attn_out + (((size_t)b * HH + hh) * TT + q0) * TT;
        for (int i = tid; i < 16 * (TT / 4); i += 256) {
            int row = i / (TT / 4);
            int c4  = i % (TT / 4);
            const float* sr = smem + row * SROW + c4 * 4;
            float4 v; v.x = sr[0]; v.y = sr[1]; v.z = sr[2]; v.w = sr[3];
            ((float4*)(outb + (size_t)row * TT))[c4] = v;
        }
    }

    // ---- O = P @ V : waves 0..3 each own one 16-wide E tile, 64 accumulating WMMAs ----
    if (wave < 4) {
        v8f acc = {};
        int ec   = wave * 16 + m;       // output / V column
        int koff = hi * 16;
        const _Float16* vcol = vbase + (size_t)ec * TT;   // contiguous along K now
        for (int k0 = 0; k0 < TT; k0 += 32) {
            v16h pa, pb;
            int kb = hi * 16;
            #pragma unroll
            for (int i = 0; i < 8; ++i) {
                pa[i]     = (_Float16)smem[m * SROW + k0 + kb + i];
                pa[8 + i] = (_Float16)smem[m * SROW + k0 + 8 + kb + i];
            }
            const _Float16* vr = vcol + k0 + koff;
            #pragma unroll
            for (int i = 0; i < 16; ++i) pb[i] = vr[i];   // 32B contiguous -> b128 loads
            acc = __builtin_amdgcn_wmma_f32_16x16x32_f16(false, pa, false, pb, (short)0, acc, false, false);
        }
        #pragma unroll
        for (int r = 0; r < 8; ++r) {
            int row = r + hi * 8;
            ctx[((size_t)b * TT + q0 + row) * HE + hh * EE + ec] = acc[r];
        }
    }
}

// ---------------- output projection out = ctx @ Wp + bp -----------------------------
__global__ void out_proj_kernel(const float* __restrict__ ctx, const float* __restrict__ Wp,
                                const float* __restrict__ bp, float* __restrict__ out) {
    int idx = blockIdx.x * blockDim.x + threadIdx.x;      // over B*T*E
    if (idx >= BB * TT * EE) return;
    int e = idx % EE;
    size_t bt = idx / EE;
    const float* c = ctx + bt * HE;
    float acc = bp[e];
    #pragma unroll 8
    for (int i = 0; i < HE; ++i) acc = fmaf(c[i], Wp[(size_t)i * EE + e], acc);
    out[idx] = acc;
}

extern "C" void kernel_launch(void* const* d_in, const int* in_sizes, int n_in,
                              void* d_out, int out_size, void* d_ws, size_t ws_size,
                              hipStream_t stream) {
    const float* x   = (const float*)d_in[0];
    const float* Wqk = (const float*)d_in[1];
    const float* bqk = (const float*)d_in[2];
    const float* Wv  = (const float*)d_in[3];
    const float* bv  = (const float*)d_in[4];
    const float* Wp  = (const float*)d_in[5];
    const float* bp  = (const float*)d_in[6];
    const unsigned char* mask = (const unsigned char*)d_in[7];

    char* ws = (char*)d_ws;
    _Float16* qh = (_Float16*)(ws);                       //  8 MB
    _Float16* kh = (_Float16*)(ws + ((size_t)8  << 20));  //  8 MB
    _Float16* vt = (_Float16*)(ws + ((size_t)16 << 20));  //  8 MB  [B,H,E,T]
    float*    ctx = (float*)  (ws + ((size_t)24 << 20));  // 16 MB

    float* out  = (float*)d_out;                          // [B,T,E]
    float* attn = out + (size_t)BB * TT * EE;             // [B,H,T,T]

    int n1 = BB * TT * 2 * HE;
    qk_conv_kernel<<<(n1 + 255) / 256, 256, 0, stream>>>(x, Wqk, bqk, qh, kh);
    int n2 = BB * TT * HE;
    value_kernel<<<(n2 + 255) / 256, 256, 0, stream>>>(x, Wv, bv, vt);

    int nblk = BB * HH * (TT / 16);                       // 4096 workgroups
    size_t smem_bytes = (size_t)16 * SROW * sizeof(float);// ~130 KB dynamic LDS (CDNA5 WGP)
    attn_kernel<<<nblk, 256, smem_bytes, stream>>>(qh, kh, vt, mask, attn, ctx);

    int n3 = BB * TT * EE;
    out_proj_kernel<<<(n3 + 255) / 256, 256, 0, stream>>>(ctx, Wp, bp, out);
}
